// S2ConvNet_BWCh_24395414241544
// MI455X (gfx1250) — matrix operand order repaired and checked by reading, not compiled, not run on hardware
//
#include <hip/hip_runtime.h>
#include <cmath>

#define DEV static __device__ __forceinline__

typedef __attribute__((ext_vector_type(16))) _Float16 v16h;
typedef __attribute__((ext_vector_type(8)))  _Float16 v8h;
typedef __attribute__((ext_vector_type(8)))  float    v8f;

#define PI_D 3.14159265358979323846
#define PI_F 3.14159265358979323846f

DEV int offMN(int l){ return l*(2*l-1)*(2*l+1)/3; }   // sum_{l'<l} (2l'+1)^2

DEV double ipow(double x, int n){ double r=1.0; for(int i=0;i<n;i++) r*=x; return r; }

// Wigner small-d, d^l_{mu,nu}(beta) = <mu| e^{-i beta Jy} |nu>, standard convention.
DEV double wigner_d(int l, int mu, int nu, double beta){
  double fact[32];
  fact[0]=1.0;
  for(int i=1;i<32;i++) fact[i]=fact[i-1]*(double)i;
  double cb = cos(0.5*beta), sb = sin(0.5*beta);
  int smin = (nu-mu) > 0 ? (nu-mu) : 0;
  int smax = (l+nu) < (l-mu) ? (l+nu) : (l-mu);
  double pref = sqrt(fact[l+mu]*fact[l-mu]*fact[l+nu]*fact[l-nu]);
  double sum = 0.0;
  for(int s=smin;s<=smax;s++){
    double term = pref / (fact[l+nu-s]*fact[s]*fact[mu-nu+s]*fact[l-mu-s]);
    term *= ipow(cb, 2*l+nu-mu-2*s) * ipow(sb, mu-nu+2*s);
    sum += ((mu-nu+s)&1) ? -term : term;
  }
  return sum;
}

DEV float dhw(int b, int j){
  double s=0.0;
  for(int l=0;l<b;l++) s += sin((2.0*j+1.0)*(2.0*l+1.0)*PI_D/(4.0*b))/(2.0*l+1.0);
  return (float)(2.0/b * sin(PI_D*(2.0*j+1.0)/(4.0*b)) * s);
}

DEV double betaIn(int k){ return PI_D*(2*k+1)/256.0; }  // B_IN=64
DEV double betaL1(int k){ return PI_D*(2*k+1)/64.0;  }  // B_L1=16
DEV double betaL2(int k){ return PI_D*(2*k+1)/32.0;  }  // B_L2=8

// ---------------- constant-generation kernels ----------------

__global__ void k_quad(float* wIn, float* wL1, float* wL2){
  int j = threadIdx.x;
  if(j<128) wIn[j]=dhw(64,j);
  if(j<32)  wL1[j]=dhw(16,j);
  if(j<16)  wL2[j]=dhw(8,j);
}

__global__ void k_dcol(float* dcol){                    // [k*256 + l^2 + mp]
  int l = blockIdx.x; int w = 2*l+1;
  for(int t=threadIdx.x; t<128*w; t+=blockDim.x){
    int k=t/w, mp=t%w;
    dcol[k*256 + l*l + mp] = (float)wigner_d(l, mp-l, 0, betaIn(k));
  }
}

__global__ void k_dl1(float* d){                        // [k*5456 + cat]
  int l=blockIdx.x; int w=2*l+1; int o=offMN(l);
  for(int t=threadIdx.x; t<32*w*w; t+=blockDim.x){
    int k=t/(w*w), r=t%(w*w);
    d[k*5456 + o + r] = (float)wigner_d(l, r/w - l, r%w - l, betaL1(k));
  }
}

__global__ void k_dl2(float* d){                        // [k*680 + cat]
  int l=blockIdx.x; int w=2*l+1; int o=offMN(l);
  for(int t=threadIdx.x; t<16*w*w; t+=blockDim.x){
    int k=t/(w*w), r=t%(w*w);
    d[k*680 + o + r] = (float)wigner_d(l, r/w - l, r%w - l, betaL2(k));
  }
}

__global__ void k_psis2(float* pre, float* pim){        // conj(PSI_S2), [(l^2+mp)*24+g]
  int l=blockIdx.x; int w=2*l+1;
  for(int t=threadIdx.x; t<w*24; t+=blockDim.x){
    int mp=t/24, g=t%24;
    double beta=(g/8+1)*(PI_D/8.0)/3.0, alpha=2.0*PI_D*(g%8)/8.0;
    double dv=wigner_d(l,mp-l,0,beta);
    double ang=(double)(mp-l)*alpha;    // conj of e^{-i m a}
    pre[(l*l+mp)*24+g]=(float)(dv*cos(ang));
    pim[(l*l+mp)*24+g]=(float)(dv*sin(ang));
  }
}

__global__ void k_psiso3(_Float16* are, _Float16* aim){ // conj(PSI_SO3) f16, row=offMN(l)+mp*w+np
  int l=blockIdx.x; int w=2*l+1; int o=offMN(l);
  for(int t=threadIdx.x; t<w*w*192; t+=blockDim.x){
    int r=t/192, g=t%192;
    int mp=r/w, np=r%w;
    int bi=g/64, ai=(g/8)%8, ci=g%8;
    double beta=(bi+1)*(PI_D/8.0)/3.0;
    double alpha=2.0*PI_D*ai/8.0;
    double gamma=2.0*PI_D*ci/8.0 - alpha;
    double dv=wigner_d(l,mp-l,np-l,beta);
    double ang=(double)(mp-l)*alpha+(double)(np-l)*gamma; // conj
    are[(o+r)*192+g]=(_Float16)(dv*cos(ang));
    aim[(o+r)*192+g]=(_Float16)(dv*sin(ang));
  }
  if(l==0){ // zero pad rows 680..687
    for(int t=threadIdx.x; t<8*192; t+=blockDim.x){
      are[680*192+t]=(_Float16)0.f; aim[680*192+t]=(_Float16)0.f;
    }
  }
}

__global__ void k_w2h(const float* w2, _Float16* w2h, int n){
  int i=blockIdx.x*blockDim.x+threadIdx.x;
  if(i<n) w2h[i]=(_Float16)w2[i];
}

// ---------------- forward pipeline ----------------

// S2 FFT: Fx[cat=l^2+mp][b]  (complex, f=1)
__global__ void __launch_bounds__(128) k_s2fft(const float* x, const float* wIn,
                                               const float* dcol, float* fxr, float* fxi){
  __shared__ float twc[128], tws[128];
  __shared__ float Xr[128][31], Xi[128][31];
  int b=blockIdx.x, tid=threadIdx.x;
  twc[tid]=cosf(2.f*PI_F*(float)tid/128.f);
  tws[tid]=sinf(2.f*PI_F*(float)tid/128.f);
  __syncthreads();
  { // direct DFT of row k at the 31 needed bins m=-15..15
    int k=tid;
    float acr[31], aci[31];
    #pragma unroll
    for(int m=0;m<31;m++){ acr[m]=0.f; aci[m]=0.f; }
    const float* row = x + b*16384 + k*128;
    for(int j=0;j<128;j++){
      float v=row[j];
      #pragma unroll
      for(int mi=0;mi<31;mi++){
        int m=mi-15;
        int h=(((-m*j)%128)+128)&127;   // e^{-2pi i m j /128}
        acr[mi]+=v*twc[h]; aci[mi]+=v*tws[h];
      }
    }
    #pragma unroll
    for(int mi=0;mi<31;mi++){ Xr[k][mi]=acr[mi]; Xi[k][mi]=aci[mi]; }
  }
  __syncthreads();
  for(int cat=tid; cat<256; cat+=128){
    int l=0; while((l+1)*(l+1)<=cat) l++;
    int mp=cat-l*l, m=mp-l;
    float sr=0.f, si=0.f;
    for(int k=0;k<128;k++){
      float c=wIn[k]*dcol[k*256+cat];
      sr+=c*Xr[k][m+15]; si+=c*Xi[k][m+15];
    }
    float sc=2.f*PI_F/128.f;
    fxr[cat*16+b]=sr*sc; fxi[cat*16+b]=si*sc;
  }
}

// FyC1[(l^2+np)*64+o] = sum_g w1[o,g]*conj(psiS2)   (conj already baked into psi tables)
__global__ void k_fy1(const float* w1, const float* pre, const float* pim,
                      float* fyr, float* fyi){
  int idx=blockIdx.x*blockDim.x+threadIdx.x;
  if(idx>=16384) return;
  int cat=idx>>6, o=idx&63;
  float sr=0.f, si=0.f;
  for(int g=0;g<24;g++){
    float w=w1[o*24+g];
    sr+=w*pre[cat*24+g]; si+=w*pim[cat*24+g];
  }
  fyr[idx]=sr; fyi[idx]=si;
}

// Fz[bf*5456 + mn] = Fx[l,mp,b] * FyC1[l,np,o]  (outer product per l)
__global__ void k_fz(const float* fxr,const float* fxi,const float* fyr,const float* fyi,
                     float* fzr, float* fzi){
  int bf=blockIdx.x; int b=bf>>6, o=bf&63;
  for(int mn=threadIdx.x; mn<5456; mn+=blockDim.x){
    int l=0; while(offMN(l+1)<=mn) l++;
    int r=mn-offMN(l), w=2*l+1;
    int mp=r/w, np=r%w;
    float ar=fxr[(l*l+mp)*16+b], ai=fxi[(l*l+mp)*16+b];
    float br=fyr[(l*l+np)*64+o], bi=fyi[(l*l+np)*64+o];
    fzr[bf*5456+mn]=ar*br-ai*bi;
    fzi[bf*5456+mn]=ar*bi+ai*br;
  }
}

// Fused SO3 inverse FFT (32^3) + ReLU + forward FFT at 15x15 bins. One block per (bf,k).
// Never materializes y1 (saves 134MB + a round trip to memory).
__global__ void __launch_bounds__(256) k_sconv_mid(const float* fzr, const float* fzi,
                                                   const float* dl1, float* xr, float* xi){
  __shared__ float Sr[32][32], Si[32][32], Tr[32][32], Ti[32][32];
  __shared__ float twc[32], tws[32];
  int blk=blockIdx.x; int k=blk&31, bf=blk>>5;
  int tid=threadIdx.x;
  if(tid<32){ twc[tid]=cosf(2.f*PI_F*(float)tid/32.f); tws[tid]=sinf(2.f*PI_F*(float)tid/32.f); }
  for(int p=tid;p<1024;p+=256){ Sr[p>>5][p&31]=0.f; Si[p>>5][p&31]=0.f; }
  __syncthreads();
  const float* Dk = dl1 + k*5456;
  const float* Fr = fzr + (long)bf*5456;
  const float* Fi = fzi + (long)bf*5456;
  for(int c=tid;c<961;c+=256){                 // assemble spectral tile
    int m=c/31-15, n=c%31-15;
    int am=m<0?-m:m, an=n<0?-n:n;
    int lmin=am>an?am:an;
    float sr=0.f, si=0.f;
    for(int l=lmin;l<16;l++){
      int idx=offMN(l)+(m+l)*(2*l+1)+(n+l);
      float cf=(float)(2*l+1)*Dk[idx];
      sr+=cf*Fr[idx]; si+=cf*Fi[idx];
    }
    Sr[m&31][n&31]=sr; Si[m&31][n&31]=si;
  }
  __syncthreads();
  for(int p=tid;p<1024;p+=256){                // inverse DFT over n
    int m=p>>5, j=p&31;
    float tr=0.f, ti=0.f;
    for(int n=0;n<32;n++){
      int h=(n*j)&31;
      float c=twc[h], s=tws[h], a=Sr[m][n], bb=Si[m][n];
      tr+=a*c-bb*s; ti+=a*s+bb*c;
    }
    Tr[m][j]=tr; Ti[m][j]=ti;
  }
  __syncthreads();
  for(int p=tid;p<1024;p+=256){                // inverse DFT over m, Re, ReLU -> Y (reuse Sr)
    int i=p>>5, j=p&31;
    float y=0.f;
    for(int m=0;m<32;m++){
      int h=(m*i)&31;
      y+=Tr[m][j]*twc[h]-Ti[m][j]*tws[h];
    }
    Sr[i][j]=y>0.f?y:0.f;
  }
  __syncthreads();
  for(int p=tid;p<480;p+=256){                 // forward DFT over i at m'=-7..7
    int mp=p/32, j=p%32; int m=mp-7;
    float ur=0.f, ui=0.f;
    for(int i=0;i<32;i++){
      int h=(((-m*i)%32)+32)&31;               // e^{-2pi i m' i/32}
      ur+=Sr[i][j]*twc[h]; ui+=Sr[i][j]*tws[h];
    }
    Tr[mp][j]=ur; Ti[mp][j]=ui;
  }
  __syncthreads();
  for(int p=tid;p<225;p+=256){                 // forward DFT over j at n'=-7..7, scale 1/32^2
    int mp=p/15, np=p%15; int n=np-7;
    float zr=0.f, zi=0.f;
    for(int j=0;j<32;j++){
      int h=(((-n*j)%32)+32)&31;
      float c=twc[h], s=tws[h], ur=Tr[mp][j], ui=Ti[mp][j];
      zr+=ur*c-ui*s; zi+=ur*s+ui*c;
    }
    xr[(long)blk*225+p]=zr*(1.f/1024.f);
    xi[(long)blk*225+p]=zi*(1.f/1024.f);
  }
}

// Fx2 -> GEMM A operand (f16), rows (mp*16+b), cols (np*64+f), per-l block
__global__ void k_fx2(const float* xr, const float* xi, const float* wL1,
                      const float* dl1, _Float16* a2r, _Float16* a2i){
  int idx=blockIdx.x*blockDim.x+threadIdx.x;
  if(idx>=680*1024) return;
  int mn=idx>>10, bf=idx&1023;
  int b=bf>>6, f=bf&63;
  int l=0; while(offMN(l+1)<=mn) l++;
  int r=mn-offMN(l), w=2*l+1;
  int mp=r/w, np=r%w;
  int bin=(mp-l+7)*15+(np-l+7);
  float sr=0.f, si=0.f;
  for(int k=0;k<32;k++){
    float c=wL1[k]*dl1[k*5456+mn];
    long q=((long)bf*32+k)*225+bin;
    sr+=c*xr[q]; si+=c*xi[q];
  }
  float tp=2.f*PI_F;
  int lda=64*w;
  long off=1024L*offMN(l);
  long addr=off+(long)(mp*16+b)*lda+np*64+f;
  a2r[addr]=(_Float16)(sr*tp);
  a2i[addr]=(_Float16)(si*tp);
}

// ---------------- WMMA helpers ----------------

DEV v16h load_a_frag(const _Float16* p, int hi){   // p = base + row*lda + kk
  const v8h lo = *(const v8h*)(p + 16*hi);
  const v8h hh = *(const v8h*)(p + 16*hi + 8);
  v16h a;
  #pragma unroll
  for(int e=0;e<8;e++){ a[e]=lo[e]; a[e+8]=hh[e]; }
  return a;
}

DEV v8f wmma16(v16h a, v16h b, v8f c){
  return __builtin_amdgcn_wmma_f32_16x16x32_f16(false, a, false, b, (short)0, c, false, false);
}

// GEMM 1: FyC(688 x 8192) = conj(PSI_SO3)(688 x 192, f16) x w2^T(192 x 8192, f16)
// Output stored as [row][o*64+i] so GEMM-2 B-fragments are contiguous v16h loads.
__global__ void __launch_bounds__(32) k_gemm_fy2(const _Float16* par, const _Float16* pai,
                                                 const _Float16* w2h,
                                                 _Float16* fcr, _Float16* fci){
  int lane=threadIdx.x;
  int row0=blockIdx.x*16, col0=blockIdx.y*16;
  int mrow=lane&15, hi=lane>>4;
  v8f ar={0.f,0.f,0.f,0.f,0.f,0.f,0.f,0.f}, ai=ar;
  const _Float16* pa_r = par + (long)(row0+mrow)*192;
  const _Float16* pa_i = pai + (long)(row0+mrow)*192;
  const _Float16* pb   = w2h + (long)(col0+mrow)*192 + 16*hi;
  #pragma unroll
  for(int kk=0;kk<192;kk+=32){
    __builtin_prefetch(pb + kk + 32, 0, 0);
    v16h fa_r=load_a_frag(pa_r+kk, hi);
    v16h fa_i=load_a_frag(pa_i+kk, hi);
    v16h fb = *(const v16h*)(pb + kk);
    ar=wmma16(fa_r,fb,ar);
    ai=wmma16(fa_i,fb,ai);
  }
  int col=col0+mrow;
  long sc=(long)(col&127)*64 + (col>>7);     // swizzled: [o*64+i]
  #pragma unroll
  for(int r2=0;r2<8;r2++){
    long row=row0 + r2 + 8*hi;
    fcr[row*8192+sc]=(_Float16)ar[r2];
    fci[row*8192+sc]=(_Float16)ai[r2];
  }
}

// GEMM 2 (per l): Fz2 = A(16w x 64w) x B(64w x 128w), complex via 4 accumulators.
__global__ void __launch_bounds__(32) k_gemm_fz2(const _Float16* a2r, const _Float16* a2i,
                                                 const _Float16* fcr, const _Float16* fci,
                                                 float* zr, float* zi, int l){
  int w=2*l+1;
  int N=128*w, K=64*w;
  int lane=threadIdx.x, mrow=lane&15, hi=lane>>4;
  int row0=blockIdx.x*16, col0=blockIdx.y*16;
  long offA=1024L*offMN(l);
  long offC=2048L*offMN(l);
  long rowF0=offMN(l);
  v8f p1={0.f,0.f,0.f,0.f,0.f,0.f,0.f,0.f}, p2=p1, p3=p1, p4=p1;
  const _Float16* par = a2r + offA + (long)(row0+mrow)*K;
  const _Float16* pai = a2i + offA + (long)(row0+mrow)*K;
  int nn=col0+mrow;
  int np=nn>>7, o=nn&127;
  for(int kk=0;kk<K;kk+=32){
    v16h far=load_a_frag(par+kk,hi);
    v16h fai=load_a_frag(pai+kk,hi);
    int kbase=kk+16*hi;
    int kp=kbase>>6, ib=kbase&63;            // kp constant across fragment
    long rowF=rowF0 + np*w + kp;             // conj(Fy2)[n_out, k_contr]
    long baddr=rowF*8192 + (long)o*64 + ib;  // i contiguous -> single v16h load
    v16h fbr=*(const v16h*)(fcr + baddr);
    v16h fbi=*(const v16h*)(fci + baddr);
    p1=wmma16(far,fbr,p1);
    p2=wmma16(fai,fbi,p2);
    p3=wmma16(far,fbi,p3);
    p4=wmma16(fai,fbr,p4);
  }
  #pragma unroll
  for(int r2=0;r2<8;r2++){
    long row=row0+r2+8*hi;
    zr[offC+row*N+nn]=p1[r2]-p2[r2];         // Re = ArBr - AiBi
    zi[offC+row*N+nn]=p3[r2]+p4[r2];         // Im = ArBi + AiBr
  }
}

// Fused SO3 inverse FFT (16^3) + ReLU + mean over (i,j). One block per (b,o,k).
__global__ void __launch_bounds__(256) k_ifft2_mean(const float* zr, const float* zi,
                                                    const float* dl2, float* partial){
  __shared__ float Sr[16][16], Si[16][16], Tr[16][16], Ti[16][16];
  __shared__ float twc[16], tws[16];
  __shared__ float red[256];
  int blk=blockIdx.x; int k=blk&15, bo=blk>>4;
  int b=bo>>7, o=bo&127;
  int tid=threadIdx.x;
  if(tid<16){ twc[tid]=cosf(2.f*PI_F*(float)tid/16.f); tws[tid]=sinf(2.f*PI_F*(float)tid/16.f); }
  { int mm=tid>>4, nn2=tid&15; Sr[mm][nn2]=0.f; Si[mm][nn2]=0.f; }
  __syncthreads();
  const float* Dk=dl2 + k*680;
  if(tid<225){
    int m=tid/15-7, n=tid%15-7;
    int am=m<0?-m:m, an=n<0?-n:n;
    int lmin=am>an?am:an;
    float sr=0.f, si=0.f;
    for(int l=lmin;l<8;l++){
      int w=2*l+1;
      int cat=offMN(l)+(m+l)*w+(n+l);
      float cf=(float)w*Dk[cat];
      long q=2048L*offMN(l)+(long)((m+l)*16+b)*(128*w)+(long)(n+l)*128+o;
      sr+=cf*zr[q]; si+=cf*zi[q];
    }
    Sr[m&15][n&15]=sr; Si[m&15][n&15]=si;
  }
  __syncthreads();
  { int mm=tid>>4, j=tid&15;
    float tr=0.f, ti2=0.f;
    for(int n=0;n<16;n++){
      int h=(n*j)&15;
      float c=twc[h], s=tws[h];
      tr+=Sr[mm][n]*c-Si[mm][n]*s;
      ti2+=Sr[mm][n]*s+Si[mm][n]*c;
    }
    Tr[mm][j]=tr; Ti[mm][j]=ti2;
  }
  __syncthreads();
  { int i=tid>>4, j=tid&15;
    float y=0.f;
    for(int m=0;m<16;m++){
      int h=(m*i)&15;
      y+=Tr[m][j]*twc[h]-Ti[m][j]*tws[h];
    }
    red[tid]=y>0.f?y:0.f;
  }
  __syncthreads();
  for(int s=128;s>0;s>>=1){
    if(tid<s) red[tid]+=red[tid+s];
    __syncthreads();
  }
  if(tid==0) partial[bo*16+k]=red[0]*(1.f/256.f);
}

// feat = sum_k W_L2[k]*partial; out = feat @ lin_w^T + lin_b
__global__ void k_head(const float* partial, const float* wL2,
                       const float* lw, const float* lb, float* out){
  __shared__ float feat[2048];
  int tid=threadIdx.x;
  for(int idx=tid; idx<2048; idx+=256){
    float s=0.f;
    for(int k=0;k<16;k++) s+=wL2[k]*partial[idx*16+k];
    feat[idx]=s;
  }
  __syncthreads();
  if(tid<160){
    int b=tid/10, c=tid%10;
    float s=lb[c];
    for(int f=0;f<128;f++) s+=lw[c*128+f]*feat[b*128+f];
    out[b*10+c]=s;
  }
}

// ---------------- host ----------------

extern "C" void kernel_launch(void* const* d_in, const int* in_sizes, int n_in,
                              void* d_out, int out_size, void* d_ws, size_t ws_size,
                              hipStream_t stream){
  const float* x  =(const float*)d_in[0];
  const float* w1 =(const float*)d_in[1];
  const float* w2 =(const float*)d_in[2];
  const float* lw =(const float*)d_in[3];
  const float* lb =(const float*)d_in[4];
  float* out=(float*)d_out;

  char* p=(char*)d_ws;
  auto alloc=[&](size_t bytes)->void*{ void* r=(void*)p; p+=(bytes+255)&~(size_t)255; return r; };

  float* wIn =(float*)alloc(128*4);
  float* wL1 =(float*)alloc(32*4);
  float* wL2 =(float*)alloc(16*4);
  float* dcol=(float*)alloc(128*256*4);
  float* dl1 =(float*)alloc((size_t)32*5456*4);
  float* dl2 =(float*)alloc((size_t)16*680*4);
  float* p2r =(float*)alloc(6144*4);
  float* p2i =(float*)alloc(6144*4);
  _Float16* par=(_Float16*)alloc((size_t)688*192*2);
  _Float16* pai=(_Float16*)alloc((size_t)688*192*2);
  _Float16* w2h=(_Float16*)alloc((size_t)1572864*2);
  float* fxr=(float*)alloc(4096*4);
  float* fxi=(float*)alloc(4096*4);
  float* fy1r=(float*)alloc(16384*4);
  float* fy1i=(float*)alloc(16384*4);
  float* fzr=(float*)alloc((size_t)5586944*4);
  float* fzi=(float*)alloc((size_t)5586944*4);
  float* xfr=(float*)alloc((size_t)7372800*4);
  float* xfi=(float*)alloc((size_t)7372800*4);
  _Float16* a2r=(_Float16*)alloc((size_t)696320*2);
  _Float16* a2i=(_Float16*)alloc((size_t)696320*2);
  _Float16* fcr=(_Float16*)alloc((size_t)688*8192*2);
  _Float16* fci=(_Float16*)alloc((size_t)688*8192*2);
  float* z2r=(float*)alloc((size_t)1392640*4);
  float* z2i=(float*)alloc((size_t)1392640*4);
  float* partial=(float*)alloc((size_t)32768*4);

  // constants (regenerated each call; deterministic, capture-safe)
  k_quad  <<<1,  128,0,stream>>>(wIn,wL1,wL2);
  k_dcol  <<<16, 128,0,stream>>>(dcol);
  k_dl1   <<<16, 256,0,stream>>>(dl1);
  k_dl2   <<<8,  256,0,stream>>>(dl2);
  k_psis2 <<<16, 128,0,stream>>>(p2r,p2i);
  k_psiso3<<<8,  256,0,stream>>>(par,pai);
  k_w2h   <<<(1572864+255)/256,256,0,stream>>>(w2,w2h,1572864);

  // forward
  k_s2fft <<<16, 128,0,stream>>>(x,wIn,dcol,fxr,fxi);
  k_fy1   <<<64, 256,0,stream>>>(w1,p2r,p2i,fy1r,fy1i);
  k_fz    <<<1024,256,0,stream>>>(fxr,fxi,fy1r,fy1i,fzr,fzi);
  k_sconv_mid<<<32768,256,0,stream>>>(fzr,fzi,dl1,xfr,xfi);
  k_fx2   <<<(680*1024+255)/256,256,0,stream>>>(xfr,xfi,wL1,dl1,a2r,a2i);
  k_gemm_fy2<<<dim3(43,512),32,0,stream>>>(par,pai,w2h,fcr,fci);
  for(int l=0;l<8;l++)
    k_gemm_fz2<<<dim3(2*l+1,8*(2*l+1)),32,0,stream>>>(a2r,a2i,fcr,fci,z2r,z2i,l);
  k_ifft2_mean<<<32768,256,0,stream>>>(z2r,z2i,dl2,partial);
  k_head  <<<1,256,0,stream>>>(partial,wL2,lw,lb,out);

  (void)in_sizes;(void)n_in;(void)out_size;(void)ws_size;
}
